// Segmentation_4947802325540
// MI455X (gfx1250) — matrix operand order repaired and checked by reading, not compiled, not run on hardware
//
#include <hip/hip_runtime.h>

typedef _Float16 h8v  __attribute__((ext_vector_type(8)));
typedef _Float16 h16v __attribute__((ext_vector_type(16)));
typedef float    f8v  __attribute__((ext_vector_type(8)));

#define DEVINL static __device__ __forceinline__

// ---------------------------------------------------------------------------
// Fragment helpers (wave32; A/B K order contiguous, C/D per CDNA5 ISA §7.12.2)
//   A: lane holds row m=lane%16, K = kc+16*half .. +15 (same order as B)
//   B: lane holds col n=lane%16, K = kc+16*half .. +15
//   C/D: lane holds col n=lane%16, rows co = 8*half + v  (v = VGPR index)
// Global K linearization for convs: k = tap*Cin + c  (c innermost == NHWC)
// ---------------------------------------------------------------------------
namespace seg {

constexpr int NB = 32, HH = 256, WW = 256, FC = 16;

DEVINL h8v h8zero() { h8v z = {0,0,0,0,0,0,0,0}; return z; }

DEVINL h16v cat16(h8v a, h8v b) {
  return __builtin_shufflevector(a, b, 0,1,2,3,4,5,6,7,8,9,10,11,12,13,14,15);
}

DEVINL h16v load_a_frag(const _Float16* __restrict__ wrow, int kb) {
  h8v a0 = *(const h8v*)(wrow + kb);
  h8v a1 = *(const h8v*)(wrow + kb + 8);
  return cat16(a0, a1);
}

DEVINL f8v wmma(h16v a, h8v b0, h8v b1, f8v acc) {
  return __builtin_amdgcn_wmma_f32_16x16x32_f16(false, a, false, cat16(b0, b1),
                                                (short)0, acc, false, false);
}

// MODE 0: conv pad=1 stride s.  MODE 1: ConvT(k3,s2,p1,op1) as lhs-dilated.
template<int MODE>
DEVINL bool coord(int h, int w, int ky, int kx, int Hin, int Win, int stride,
                  int& iy, int& ix) {
  if (MODE == 0) {
    iy = h * stride + ky - 1;
    ix = w * stride + kx - 1;
    return (unsigned)iy < (unsigned)Hin && (unsigned)ix < (unsigned)Win;
  } else {
    int qy = h + ky - 1, qx = w + kx - 1;
    iy = qy >> 1; ix = qx >> 1;
    return qy >= 0 && qx >= 0 && !(qy & 1) && !(qx & 1) && iy < Hin && ix < Win;
  }
}

// 8-channel contiguous run from the (possibly concatenated) NHWC input.
template<int C0, int C1>
DEVINL h8v ldrun(const _Float16* __restrict__ in0, const _Float16* __restrict__ in1,
                 size_t pix, int c) {
  if constexpr (C1 == 0) {
    return *(const h8v*)(in0 + pix * C0 + c);
  } else {
    return (c < C0) ? *(const h8v*)(in0 + pix * C0 + c)
                    : *(const h8v*)(in1 + pix * C1 + (c - C0));
  }
}

// First layer (Cin==1): k == tap, scalar 9-tap gather.
template<int MODE>
DEVINL h8v load_b_c1(const _Float16* __restrict__ in0, int n, int h, int w,
                     int Hin, int Win, int stride, int kb) {
  h8v r = h8zero();
#pragma unroll
  for (int e = 0; e < 8; ++e) {
    int tap = kb + e;
    if (tap < 9) {
      int ky = tap / 3, kx = tap % 3, iy, ix;
      if (coord<MODE>(h, w, ky, kx, Hin, Win, stride, iy, ix))
        r[e] = in0[(size_t)(n * Hin + iy) * Win + ix];
    }
  }
  return r;
}

// Cin==16: one 32-K chunk spans tap pair; lane half picks the tap.
template<int MODE>
DEVINL void load_tap16(const _Float16* __restrict__ in0, int n, int h, int w,
                       int Hin, int Win, int stride, int tap, h8v& b0, h8v& b1) {
  if (tap >= 9) { b0 = h8zero(); b1 = h8zero(); return; }
  int ky = (tap * 11) >> 5;            // tap/3 for tap in [0,9]
  int kx = tap - 3 * ky;
  int iy, ix;
  if (!coord<MODE>(h, w, ky, kx, Hin, Win, stride, iy, ix)) {
    b0 = h8zero(); b1 = h8zero(); return;
  }
  const _Float16* p = in0 + ((size_t)(n * Hin + iy) * Win + ix) * 16;
  b0 = *(const h8v*)p;
  b1 = *(const h8v*)(p + 8);
}

// Core: one wave computes 16(Cout) x 32(pixels) = two C tiles, A reused.
template<int MODE, int C0, int C1>
DEVINL void gemm2(const _Float16* __restrict__ in0,
                  const _Float16* __restrict__ in1,
                  const _Float16* __restrict__ wrow,
                  int n, int h, int wcol0, int wcol1,
                  int Hin, int Win, int stride,
                  f8v& acc0, f8v& acc1) {
  constexpr int CIN = C0 + C1;
  int half = (threadIdx.x & 31) >> 4;

  if constexpr (CIN == 1) {            // KPAD = 32, single chunk
    h16v a = load_a_frag(wrow, half * 16);
    int kb = half * 16;
    h8v b00 = load_b_c1<MODE>(in0, n, h, wcol0, Hin, Win, stride, kb);
    h8v b01 = load_b_c1<MODE>(in0, n, h, wcol0, Hin, Win, stride, kb + 8);
    h8v b10 = load_b_c1<MODE>(in0, n, h, wcol1, Hin, Win, stride, kb);
    h8v b11 = load_b_c1<MODE>(in0, n, h, wcol1, Hin, Win, stride, kb + 8);
    acc0 = wmma(a, b00, b01, acc0);
    acc1 = wmma(a, b10, b11, acc1);
  } else if constexpr (CIN == 16) {    // KPAD = 160, 5 chunks, tap = kc/16+half
#pragma unroll
    for (int kc = 0; kc < 160; kc += 32) {
      h16v a = load_a_frag(wrow, kc + half * 16);
      int tap = (kc >> 4) + half;
      h8v b00, b01, b10, b11;
      load_tap16<MODE>(in0, n, h, wcol0, Hin, Win, stride, tap, b00, b01);
      load_tap16<MODE>(in0, n, h, wcol1, Hin, Win, stride, tap, b10, b11);
      acc0 = wmma(a, b00, b01, acc0);
      acc1 = wmma(a, b10, b11, acc1);
    }
  } else {                             // CIN >= 32: taps outer, channels inner
#pragma unroll 1
    for (int tap = 0; tap < 9; ++tap) {
      int ky = tap / 3;
      int kx = tap - 3 * ky;
      int iy, ix0, ix1;
      bool oky, okx0, okx1;
      if constexpr (MODE == 0) {
        iy  = h * stride + ky - 1;       oky  = (unsigned)iy  < (unsigned)Hin;
        ix0 = wcol0 * stride + kx - 1;   okx0 = (unsigned)ix0 < (unsigned)Win;
        ix1 = wcol1 * stride + kx - 1;   okx1 = (unsigned)ix1 < (unsigned)Win;
      } else {
        int qy = h + ky - 1;      iy  = qy >> 1;
        oky  = qy >= 0 && !(qy & 1) && iy < Hin;
        int qx0 = wcol0 + kx - 1; ix0 = qx0 >> 1;
        okx0 = qx0 >= 0 && !(qx0 & 1) && ix0 < Win;
        int qx1 = wcol1 + kx - 1; ix1 = qx1 >> 1;
        okx1 = qx1 >= 0 && !(qx1 & 1) && ix1 < Win;
      }
      bool okA = oky && okx0, okB = oky && okx1;
      size_t rowb = (size_t)(n * Hin + iy) * Win;
      size_t pixA = rowb + ix0, pixB = rowb + ix1;
      const _Float16* wtap = wrow + tap * CIN + half * 16;
#pragma unroll
      for (int c0 = 0; c0 < CIN; c0 += 32) {
        h16v a = load_a_frag(wtap, c0);
        int c = c0 + (half << 4);
        h8v b00 = okA ? ldrun<C0, C1>(in0, in1, pixA, c)     : h8zero();
        h8v b01 = okA ? ldrun<C0, C1>(in0, in1, pixA, c + 8) : h8zero();
        h8v b10 = okB ? ldrun<C0, C1>(in0, in1, pixB, c)     : h8zero();
        h8v b11 = okB ? ldrun<C0, C1>(in0, in1, pixB, c + 8) : h8zero();
        acc0 = wmma(a, b00, b01, acc0);
        acc1 = wmma(a, b10, b11, acc1);
      }
    }
  }
}

} // namespace seg

// ---------------------------------------------------------------------------
// Kernels
// ---------------------------------------------------------------------------

// Generic conv / transposed-conv, bias+ReLU, fp16 NHWC in/out.
template<int MODE, int C0, int C1, int COUT>
__global__ void __launch_bounds__(256)
seg_conv(const _Float16* __restrict__ in0, const _Float16* __restrict__ in1,
         const _Float16* __restrict__ wT, const float* __restrict__ bias,
         _Float16* __restrict__ out,
         int Hin, int Win, int Hout, int Wout, int stride, int ntiles) {
  constexpr int CIN  = C0 + C1;
  constexpr int KPAD = (9 * CIN + 31) / 32 * 32;
  int wid = blockIdx.x * (blockDim.x >> 5) + (threadIdx.x >> 5);
  if (wid >= ntiles) return;            // wave-uniform
  int tw = Wout >> 5;                   // 32-pixel tiles
  int t = wid;
  int wbase  = (t % tw) << 5;          t /= tw;
  int h      =  t % Hout;              t /= Hout;
  int cobase = (t % (COUT >> 4)) << 4; t /= (COUT >> 4);
  int n      =  t;
  int lane = threadIdx.x & 31, half = lane >> 4, m = lane & 15;
  const _Float16* wrow = wT + (size_t)(cobase + m) * KPAD;
  __builtin_prefetch(wrow, 0, 1);       // global_prefetch_b8
  f8v acc0 = {0.f,0.f,0.f,0.f,0.f,0.f,0.f,0.f};
  f8v acc1 = {0.f,0.f,0.f,0.f,0.f,0.f,0.f,0.f};
  seg::gemm2<MODE, C0, C1>(in0, in1, wrow, n, h, wbase + m, wbase + 16 + m,
                           Hin, Win, stride, acc0, acc1);
  int co0 = cobase + (half << 3);
  h8v r0, r1;
#pragma unroll
  for (int v = 0; v < 8; ++v) {
    float b = bias[co0 + v];
    r0[v] = (_Float16)fmaxf(acc0[v] + b, 0.f);
    r1[v] = (_Float16)fmaxf(acc1[v] + b, 0.f);
  }
  size_t rowo = (size_t)(n * Hout + h) * Wout;
  *(h8v*)(out + (rowo + wbase + m)      * COUT + co0) = r0;
  *(h8v*)(out + (rowo + wbase + 16 + m) * COUT + co0) = r1;
}

// KNN aggregation: conv(u3[idx[n,k]]) + ReLU, mean over K=3 neighbors.
__global__ void __launch_bounds__(256)
seg_conv_knn(const _Float16* __restrict__ u3, const _Float16* __restrict__ wT,
             const float* __restrict__ bias, const int* __restrict__ knn_idx,
             _Float16* __restrict__ agg, int ntiles) {
  using namespace seg;
  int wid = blockIdx.x * (blockDim.x >> 5) + (threadIdx.x >> 5);
  if (wid >= ntiles) return;
  int tw = WW >> 5;
  int t = wid;
  int wbase = (t % tw) << 5; t /= tw;
  int h = t % HH;            t /= HH;
  int n = t;                            // Cout/16 == 1
  int lane = threadIdx.x & 31, half = lane >> 4, m = lane & 15;
  const _Float16* wrow = wT + (size_t)m * 160;
  __builtin_prefetch(wrow, 0, 1);
  int co0 = half << 3;
  float ms0[8] = {0,0,0,0,0,0,0,0};
  float ms1[8] = {0,0,0,0,0,0,0,0};
#pragma unroll 1
  for (int kk = 0; kk < 3; ++kk) {
    int nsrc = knn_idx[n * 3 + kk];
    f8v acc0 = {0.f,0.f,0.f,0.f,0.f,0.f,0.f,0.f};
    f8v acc1 = {0.f,0.f,0.f,0.f,0.f,0.f,0.f,0.f};
    gemm2<0, 16, 0>(u3, nullptr, wrow, nsrc, h, wbase + m, wbase + 16 + m,
                    HH, WW, 1, acc0, acc1);
#pragma unroll
    for (int v = 0; v < 8; ++v) {
      float b = bias[co0 + v];
      ms0[v] += fmaxf(acc0[v] + b, 0.f);
      ms1[v] += fmaxf(acc1[v] + b, 0.f);
    }
  }
  h8v r0, r1;
#pragma unroll
  for (int v = 0; v < 8; ++v) {
    r0[v] = (_Float16)(ms0[v] * (1.f / 3.f));
    r1[v] = (_Float16)(ms1[v] * (1.f / 3.f));
  }
  size_t rowo = (size_t)(n * HH + h) * WW;
  *(h8v*)(agg + (rowo + wbase + m)      * FC + co0) = r0;
  *(h8v*)(agg + (rowo + wbase + 16 + m) * FC + co0) = r1;
}

// Gram matrix: sim[32][32] = feats @ feats^T via WMMA, split over K.
__global__ void __launch_bounds__(256)
seg_sim_gemm(const _Float16* __restrict__ feats, float* __restrict__ sim,
             int parts) {
  using namespace seg;
  const int Kf = HH * WW * FC;                  // 1,048,576 per slice
  int wid = blockIdx.x * (blockDim.x >> 5) + (threadIdx.x >> 5);
  int combo = wid & 3;
  int kp = wid >> 2;
  if (kp >= parts) return;
  int i0 = (combo >> 1) << 4, j0 = (combo & 1) << 4;
  int kslice = Kf / parts;
  int k0 = kp * kslice;
  int lane = threadIdx.x & 31, half = lane >> 4, m = lane & 15;
  const _Float16* arow = feats + (size_t)(i0 + m) * Kf;
  const _Float16* brow = feats + (size_t)(j0 + m) * Kf;
  f8v acc = {0.f,0.f,0.f,0.f,0.f,0.f,0.f,0.f};
  for (int kc = k0; kc < k0 + kslice; kc += 32) {
    h16v a = load_a_frag(arow, kc + half * 16);
    int kb = kc + half * 16;
    h8v b0 = *(const h8v*)(brow + kb);
    h8v b1 = *(const h8v*)(brow + kb + 8);
    acc = wmma(a, b0, b1, acc);
  }
#pragma unroll
  for (int v = 0; v < 8; ++v)
    atomicAdd(&sim[(i0 + v + (half << 3)) * 32 + (j0 + m)], acc[v]);
}

// Normalize by norms (from diagonal), mask groups/self, take top-3.
__global__ void seg_topk(const float* __restrict__ sim,
                         const int* __restrict__ prange,
                         int* __restrict__ knn_idx) {
  __shared__ float norms[32];
  int i = threadIdx.x;
  norms[i] = fmaxf(sqrtf(fmaxf(sim[i * 33], 0.f)), 1e-8f);
  __syncthreads();
  int gi = (i >= prange[0]) + (i >= prange[1]);
  float bv0 = -3.4e38f, bv1 = -3.4e38f, bv2 = -3.4e38f;
  int bi0 = 0, bi1 = 0, bi2 = 0;
  for (int j = 0; j < 32; ++j) {
    if (j == i) continue;
    int gj = (j >= prange[0]) + (j >= prange[1]);
    if (gj != gi) continue;
    float v = sim[i * 32 + j] / (norms[i] * norms[j]);
    if (v > bv0)      { bv2=bv1; bi2=bi1; bv1=bv0; bi1=bi0; bv0=v; bi0=j; }
    else if (v > bv1) { bv2=bv1; bi2=bi1; bv1=v; bi1=j; }
    else if (v > bv2) { bv2=v; bi2=j; }
  }
  knn_idx[i * 3 + 0] = bi0;
  knn_idx[i * 3 + 1] = bi1;
  knn_idx[i * 3 + 2] = bi2;
}

// Final 1x1 dense (32 -> 2) + softmax, NCHW f32 output.
__global__ void seg_head(const _Float16* __restrict__ gnn,
                         const _Float16* __restrict__ d1,
                         const float* __restrict__ wde,
                         const float* __restrict__ bde,
                         float* __restrict__ out) {
  const int NP = 32 * 256 * 256;
  int p = blockIdx.x * blockDim.x + threadIdx.x;
  if (p >= NP) return;
  const _Float16* g = gnn + (size_t)p * 16;
  const _Float16* d = d1  + (size_t)p * 16;
  float l0 = bde[0], l1 = bde[1];
#pragma unroll
  for (int c = 0; c < 16; ++c) {
    float f = (float)g[c]; l0 += f * wde[2 * c]; l1 += f * wde[2 * c + 1];
  }
#pragma unroll
  for (int c = 0; c < 16; ++c) {
    float f = (float)d[c]; l0 += f * wde[2 * (16 + c)]; l1 += f * wde[2 * (16 + c) + 1];
  }
  float mx = fmaxf(l0, l1);
  float e0 = __expf(l0 - mx), e1 = __expf(l1 - mx);
  float inv = 1.f / (e0 + e1);
  int n = p / (256 * 256), hw = p % (256 * 256);
  float* o = out + (size_t)n * 2 * 256 * 256 + hw;
  o[0] = e0 * inv;
  o[256 * 256] = e1 * inv;
}

// OIHW f32 -> [O][Kpad] f16, K = tap*Cin + c, zero padded past 9*Cin.
__global__ void seg_wconv(const float* __restrict__ w, _Float16* __restrict__ wt,
                          int O, int Cin, int Kpad) {
  int t = blockIdx.x * blockDim.x + threadIdx.x;
  if (t >= O * Kpad) return;
  int o = t / Kpad, k = t % Kpad;
  _Float16 v = (_Float16)0.f;
  if (k < 9 * Cin) {
    int tap = k / Cin, c = k % Cin;
    v = (_Float16)w[(o * Cin + c) * 9 + tap];
  }
  wt[t] = v;
}

__global__ void seg_cvt_in(const float* __restrict__ x, _Float16* __restrict__ y,
                           int n) {
  int i = blockIdx.x * blockDim.x + threadIdx.x;
  if (i < n) y[i] = (_Float16)x[i];
}

__global__ void seg_zero(float* __restrict__ p, int n) {
  int i = blockIdx.x * blockDim.x + threadIdx.x;
  if (i < n) p[i] = 0.f;
}

// ---------------------------------------------------------------------------
// Host orchestration
// ---------------------------------------------------------------------------
extern "C" void kernel_launch(void* const* d_in, const int* in_sizes, int n_in,
                              void* d_out, int out_size, void* d_ws, size_t ws_size,
                              hipStream_t stream) {
  (void)in_sizes; (void)n_in; (void)out_size; (void)ws_size;

  const float* cts  = (const float*)d_in[0];
  const int*   prng = (const int*)d_in[1];
  const float* w_d1 = (const float*)d_in[2];  const float* b_d1 = (const float*)d_in[3];
  const float* w_d2 = (const float*)d_in[4];  const float* b_d2 = (const float*)d_in[5];
  const float* w_d3 = (const float*)d_in[6];  const float* b_d3 = (const float*)d_in[7];
  const float* w_d4 = (const float*)d_in[8];  const float* b_d4 = (const float*)d_in[9];
  const float* w_u1 = (const float*)d_in[10]; const float* b_u1 = (const float*)d_in[11];
  const float* w_u2 = (const float*)d_in[12]; const float* b_u2 = (const float*)d_in[13];
  const float* w_u3 = (const float*)d_in[14]; const float* b_u3 = (const float*)d_in[15];
  const float* w_ga = (const float*)d_in[16]; const float* b_ga = (const float*)d_in[17];
  const float* w_gu = (const float*)d_in[18]; const float* b_gu = (const float*)d_in[19];
  const float* w_de = (const float*)d_in[20]; const float* b_de = (const float*)d_in[21];

  char* cur = (char*)d_ws;
  auto take = [&](size_t bytes) -> char* {
    char* p = cur; cur += (bytes + 255) & ~(size_t)255; return p;
  };

  const int KP1 = 32, KP16 = 160, KP32 = 288, KP64 = 576, KP128 = 1152;
  _Float16* wt_d1 = (_Float16*)take((size_t)16  * KP1   * 2);
  _Float16* wt_d2 = (_Float16*)take((size_t)32  * KP16  * 2);
  _Float16* wt_d3 = (_Float16*)take((size_t)64  * KP32  * 2);
  _Float16* wt_d4 = (_Float16*)take((size_t)128 * KP64  * 2);
  _Float16* wt_u1 = (_Float16*)take((size_t)64  * KP128 * 2);
  _Float16* wt_u2 = (_Float16*)take((size_t)32  * KP128 * 2);
  _Float16* wt_u3 = (_Float16*)take((size_t)16  * KP64  * 2);
  _Float16* wt_ga = (_Float16*)take((size_t)16  * KP16  * 2);
  _Float16* wt_gu = (_Float16*)take((size_t)16  * KP32  * 2);

  const size_t PIX = (size_t)32 * 256 * 256;          // N*H*W
  _Float16* x0   = (_Float16*)take(PIX * 1 * 2);      // NHWC fp16 activations
  _Float16* a_d1 = (_Float16*)take(PIX * 16 * 2);
  _Float16* a_d2 = (_Float16*)take(PIX / 4 * 32 * 2);
  _Float16* a_d3 = (_Float16*)take(PIX / 16 * 64 * 2);
  _Float16* a_d4 = (_Float16*)take(PIX / 64 * 128 * 2);
  _Float16* a_u1 = (_Float16*)take(PIX / 16 * 64 * 2);
  _Float16* a_u2 = (_Float16*)take(PIX / 4 * 32 * 2);
  _Float16* a_u3 = (_Float16*)take(PIX * 16 * 2);
  _Float16* a_ag = (_Float16*)take(PIX * 16 * 2);
  _Float16* a_gn = (_Float16*)take(PIX * 16 * 2);
  float*    simb = (float*)take(32 * 32 * 4);
  int*      kidx = (int*)take(32 * 3 * 4);

  // 1. Weight repack to fp16 [O][Kpad]
  auto wcv = [&](const float* w, _Float16* wt, int O, int Cin, int Kpad) {
    int n = O * Kpad;
    seg_wconv<<<dim3((n + 255) / 256), dim3(256), 0, stream>>>(w, wt, O, Cin, Kpad);
  };
  wcv(w_d1, wt_d1, 16, 1, KP1);
  wcv(w_d2, wt_d2, 32, 16, KP16);
  wcv(w_d3, wt_d3, 64, 32, KP32);
  wcv(w_d4, wt_d4, 128, 64, KP64);
  wcv(w_u1, wt_u1, 64, 128, KP128);
  wcv(w_u2, wt_u2, 32, 128, KP128);
  wcv(w_u3, wt_u3, 16, 64, KP64);
  wcv(w_ga, wt_ga, 16, 16, KP16);
  wcv(w_gu, wt_gu, 16, 32, KP32);

  // 2. Input f32 -> f16 (C=1, NCHW==NHWC)
  seg_cvt_in<<<dim3((int)((PIX + 255) / 256)), dim3(256), 0, stream>>>(cts, x0, (int)PIX);

  // 3. Conv stack (each wave: 16 Cout x 32 pixels)
#define LAUNCH_CONV(MODE, C0v, C1v, COUTv, i0, i1, wt, b, o, Hin, Win, Hout, Wout, stride)      \
  { int nt = 32 * ((COUTv) / 16) * (Hout) * ((Wout) / 32);                                      \
    seg_conv<MODE, C0v, C1v, COUTv><<<dim3((nt + 7) / 8), dim3(256), 0, stream>>>(              \
        i0, i1, wt, b, o, Hin, Win, Hout, Wout, stride, nt); }

  LAUNCH_CONV(0, 1,   0,  16,  x0,   (const _Float16*)nullptr, wt_d1, b_d1, a_d1, 256, 256, 256, 256, 1);
  LAUNCH_CONV(0, 16,  0,  32,  a_d1, (const _Float16*)nullptr, wt_d2, b_d2, a_d2, 256, 256, 128, 128, 2);
  LAUNCH_CONV(0, 32,  0,  64,  a_d2, (const _Float16*)nullptr, wt_d3, b_d3, a_d3, 128, 128, 64,  64,  2);
  LAUNCH_CONV(0, 64,  0,  128, a_d3, (const _Float16*)nullptr, wt_d4, b_d4, a_d4, 64,  64,  32,  32,  2);
  LAUNCH_CONV(1, 128, 0,  64,  a_d4, (const _Float16*)nullptr, wt_u1, b_u1, a_u1, 32,  32,  64,  64,  2);
  LAUNCH_CONV(1, 64,  64, 32,  a_u1, a_d3,                     wt_u2, b_u2, a_u2, 64,  64, 128, 128,  2);
  LAUNCH_CONV(1, 32,  32, 16,  a_u2, a_d2,                     wt_u3, b_u3, a_u3, 128, 128, 256, 256, 2);

  // 4. Gram matrix + top-3 neighbors
  seg_zero<<<dim3(4), dim3(256), 0, stream>>>(simb, 1024);
  const int parts = 512;                       // 2048 waves: 4 tiles x 512 K-parts
  seg_sim_gemm<<<dim3(parts * 4 / 8), dim3(256), 0, stream>>>(a_u3, simb, parts);
  seg_topk<<<dim3(1), dim3(32), 0, stream>>>(simb, prng, kidx);

  // 5. KNN conv + mean
  {
    int nt = 32 * 256 * (256 / 32);
    seg_conv_knn<<<dim3((nt + 7) / 8), dim3(256), 0, stream>>>(
        a_u3, wt_ga, b_ga, kidx, a_ag, nt);
  }

  // 6. Fusion conv on concat(u3, agg)
  LAUNCH_CONV(0, 16, 16, 16, a_u3, a_ag, wt_gu, b_gu, a_gn, 256, 256, 256, 256, 1);

  // 7. Dense head + softmax -> f32 NCHW output
  seg_head<<<dim3((int)((PIX + 255) / 256)), dim3(256), 0, stream>>>(
      a_gn, a_d1, w_de, b_de, (float*)d_out);
#undef LAUNCH_CONV
}
